// SpatioTemporalTransformer_35021163332120
// MI455X (gfx1250) — matrix operand order repaired and checked by reading, not compiled, not run on hardware
//
#include <hip/hip_runtime.h>

// ---------------------------------------------------------------------------
// CDNA5 (gfx1250) wave32 WMMA types
// ---------------------------------------------------------------------------
typedef __bf16 bf16_t;
typedef __attribute__((ext_vector_type(16))) __bf16 v16bf;
typedef __attribute__((ext_vector_type(8)))  float  v8f;

union FragBF { v16bf v; unsigned u[8]; int4 q[2]; };

__device__ __forceinline__ v8f wmma_bf16(const FragBF& a, const FragBF& b, v8f c) {
  return __builtin_amdgcn_wmma_f32_16x16x32_bf16(false, a.v, false, b.v,
                                                 (short)0, c, false, false);
}

__device__ __forceinline__ float allreduce_sum(float v) {
#pragma unroll
  for (int off = 16; off > 0; off >>= 1) v += __shfl_xor(v, off, 32);
  return v;
}

// Async global->LDS copy (gfx1250 GLOBAL_LOAD_ASYNC_TO_LDS_B128), guarded so the
// source still compiles on toolchains that don't declare the builtin.
#if defined(__has_builtin)
#if __has_builtin(__builtin_amdgcn_global_load_async_to_lds_b128) && \
    __has_builtin(__builtin_amdgcn_s_wait_asynccnt)
#define USE_ASYNC_LDS 1
#else
#define USE_ASYNC_LDS 0
#endif
#else
#define USE_ASYNC_LDS 0
#endif

#if USE_ASYNC_LDS
// Builtin takes typed int4 vector pointers: AS1 source, AS3 LDS dest, imm offset/cpol.
typedef int v4i_t __attribute__((ext_vector_type(4)));
typedef __attribute__((address_space(1))) v4i_t v4i_as1;
typedef __attribute__((address_space(3))) v4i_t v4i_as3;

__device__ __forceinline__ void copy16_g2l(const bf16_t* g, bf16_t* l) {
  __builtin_amdgcn_global_load_async_to_lds_b128((v4i_as1*)g, (v4i_as3*)l, 0, 0);
}
__device__ __forceinline__ void async_join() {
  __builtin_amdgcn_s_wait_asynccnt(0);
}
#else
__device__ __forceinline__ void copy16_g2l(const bf16_t* g, bf16_t* l) {
  *(int4*)l = *(const int4*)g;
}
__device__ __forceinline__ void async_join() {}
#endif

// ---------------------------------------------------------------------------
// Weight prep: fp32 [K][N] -> bf16 [N][K] (transpose -> contiguous K-runs)
// ---------------------------------------------------------------------------
__global__ void wprep_kernel(const float* __restrict__ src, bf16_t* __restrict__ dst,
                             int K, int N) {
  int idx = blockIdx.x * blockDim.x + threadIdx.x;
  if (idx >= K * N) return;
  int k = idx / N, n = idx - k * N;
  dst[(size_t)n * K + k] = (bf16_t)src[idx];
}

__global__ void copy_kernel(const float4* __restrict__ src, float4* __restrict__ dst, int n) {
  int i = blockIdx.x * blockDim.x + threadIdx.x;
  if (i < n) dst[i] = src[i];
}

// ---------------------------------------------------------------------------
// GEMM: C[M,N] = A[M,K] * Bt[N,K]^T. 128x128 WG tile, 8 waves (4M x 2N),
// K step 32, double-buffered LDS staging (async copies overlap WMMA).
// ---------------------------------------------------------------------------
template<bool OUT_F32>
__global__ __launch_bounds__(256) void gemm_kernel(
    const bf16_t* __restrict__ A, const bf16_t* __restrict__ Bt,
    float* __restrict__ Cf, bf16_t* __restrict__ Cb,
    const float* __restrict__ bias, int M, int N, int K) {
  __shared__ __align__(16) bf16_t As[2][128 * 40];
  __shared__ __align__(16) bf16_t Bs[2][128 * 40];
  const int tid = threadIdx.x;
  const int lane = tid & 31, wave = tid >> 5;
  const int lhalf = lane & 15, khi = lane >> 4;
  const int wm = wave & 3, wn = wave >> 2;
  const int m0 = blockIdx.y * 128;
  const int n0 = blockIdx.x * 128;

  // Staging assignment: 16B per thread per half-tile (rows 0..63 / 64..127)
  const int lin = tid * 8;
  const int rowL = lin >> 5, colL = lin & 31;

  auto stage = [&](int buf, int k0) {
    copy16_g2l(A  + (size_t)(m0 + rowL)      * K + k0 + colL, &As[buf][rowL * 40 + colL]);
    copy16_g2l(A  + (size_t)(m0 + rowL + 64) * K + k0 + colL, &As[buf][(rowL + 64) * 40 + colL]);
    copy16_g2l(Bt + (size_t)(n0 + rowL)      * K + k0 + colL, &Bs[buf][rowL * 40 + colL]);
    copy16_g2l(Bt + (size_t)(n0 + rowL + 64) * K + k0 + colL, &Bs[buf][(rowL + 64) * 40 + colL]);
  };

  v8f zero = {0.f, 0.f, 0.f, 0.f, 0.f, 0.f, 0.f, 0.f};
  v8f acc[2][4];
#pragma unroll
  for (int mt = 0; mt < 2; ++mt)
#pragma unroll
    for (int nt = 0; nt < 4; ++nt) acc[mt][nt] = zero;

  const int nk = K >> 5;
  stage(0, 0);
  async_join();
  __syncthreads();

  for (int it = 0; it < nk; ++it) {
    const int buf = it & 1;
    if (it + 1 < nk) stage(buf ^ 1, (it + 1) << 5);

    FragBF a[2], b[4];
    // A frag (ISA 16-bit A 16x32): per-lane-half K runs [b..b+7],[b+16..b+23]
#pragma unroll
    for (int mt = 0; mt < 2; ++mt) {
      const bf16_t* p = &As[buf][(wm * 32 + mt * 16 + lhalf) * 40 + khi * 8];
      a[mt].q[0] = *(const int4*)p;
      a[mt].q[1] = *(const int4*)(p + 16);
    }
    // B frag: lane = N column; per-lane-half K runs [b..b+7],[b+16..b+23]
#pragma unroll
    for (int nt = 0; nt < 4; ++nt) {
      const bf16_t* p = &Bs[buf][(wn * 64 + nt * 16 + lhalf) * 40 + khi * 8];
      b[nt].q[0] = *(const int4*)p;
      b[nt].q[1] = *(const int4*)(p + 16);
    }
#pragma unroll
    for (int mt = 0; mt < 2; ++mt)
#pragma unroll
      for (int nt = 0; nt < 4; ++nt)
        acc[mt][nt] = wmma_bf16(a[mt], b[nt], acc[mt][nt]);

    async_join();
    __syncthreads();
  }

  // Epilogue: C layout -> VGPR r = row r (lanes 0-15) / r+8 (lanes 16-31)
  const int rowadd = khi * 8;
#pragma unroll
  for (int mt = 0; mt < 2; ++mt) {
#pragma unroll
    for (int nt = 0; nt < 4; ++nt) {
      int col = n0 + wn * 64 + nt * 16 + lhalf;
      int rb = m0 + wm * 32 + mt * 16 + rowadd;
      float bv = OUT_F32 ? bias[col] : 0.f;
#pragma unroll
      for (int r = 0; r < 8; ++r) {
        size_t idx = (size_t)(rb + r) * N + col;
        float v = acc[mt][nt][r];
        if (OUT_F32) Cf[idx] = v + bv;
        else         Cb[idx] = (bf16_t)v;
      }
    }
  }
}

// ---------------------------------------------------------------------------
// Rotary helper: rotate 8 contiguous bf16 (4 even/odd pairs) at head-dim d0
// for token position pos. Native v_exp/v_sin/v_cos ops only.
// 10000^(-j/32) == exp2(-j * log2(10000)/32); log2(10000)/32 = 0.4152410118.
// ---------------------------------------------------------------------------
__device__ __forceinline__ void rot8(const bf16_t* __restrict__ src,
                                     bf16_t* __restrict__ dst, int pos, int d0) {
#pragma unroll
  for (int e = 0; e < 4; ++e) {
    int j = (d0 >> 1) + e;
    float freq = exp2f((float)j * -0.4152410118f);  // AMD exp unit is base-2 native
    float th = (float)pos * freq;
    float s = __sinf(th), c = __cosf(th);
    float x1 = (float)src[2 * e], x2 = (float)src[2 * e + 1];
    dst[2 * e]     = (bf16_t)(x1 * c - x2 * s);
    dst[2 * e + 1] = (bf16_t)(x2 * c + x1 * s);
  }
}

// ---------------------------------------------------------------------------
// Fused attention (rotary fused into Q/K loads): per (sequence, head).
// K rotated + V^T staged in LDS; QK^T and PV via WMMA; exact softmax with
// half-wave reductions; grid.x = nseq*8, grid.y = NCTX/(NWAVES*16).
// ---------------------------------------------------------------------------
template<int NCTX, int NWAVES>
__global__ __launch_bounds__(NWAVES * 32) void attn_kernel(
    const bf16_t* __restrict__ qkv, bf16_t* __restrict__ out) {
  extern __shared__ char smem_raw[];
  bf16_t* smem = (bf16_t*)smem_raw;
  constexpr int KSTRIDE = 72;        // 64 dh + pad (rows 16B aligned)
  constexpr int VSTRIDE = NCTX + 8;  // padded (rows 16B aligned)
  constexpr int NT = NCTX / 16;

  const int tid = threadIdx.x, lane = tid & 31, wave = tid >> 5;
  const int lhalf = lane & 15, khi = lane >> 4;
  const int sh = blockIdx.x;
  const int seq = sh >> 3, h = sh & 7;
  const size_t rowbase = (size_t)seq * NCTX;

  bf16_t* Klds = smem;                                        // [NCTX][KSTRIDE]
  bf16_t* Vlds = Klds + NCTX * KSTRIDE;                       // [64][VSTRIDE] (V^T)
  bf16_t* Plds = Vlds + 64 * VSTRIDE + wave * 16 * VSTRIDE;   // per-wave [16][VSTRIDE]

  // Stage K (with rotary) and V (transposed)
  for (int idx = tid; idx < NCTX * 8; idx += NWAVES * 32) {
    int j = idx >> 3, c = (idx & 7) << 3;
    const bf16_t* src = qkv + (rowbase + j) * 1536 + h * 64 + c;
    int4 kraw = *(const int4*)(src + 512);
    int4 krot;
    rot8((const bf16_t*)&kraw, (bf16_t*)&krot, j, c);
    *(int4*)(Klds + j * KSTRIDE + c) = krot;
    int4 vv = *(const int4*)(src + 1024);
    const bf16_t* vp = (const bf16_t*)&vv;
#pragma unroll
    for (int e = 0; e < 8; ++e) Vlds[(c + e) * VSTRIDE + j] = vp[e];
  }
  __syncthreads();

  const int i0 = blockIdx.y * (NWAVES * 16) + wave * 16;
  const int pos = i0 + lhalf;

  // Q fragments from global with fused rotary
  FragBF qf[2];
  {
    const bf16_t* qp = qkv + (rowbase + pos) * 1536 + h * 64;
#pragma unroll
    for (int dt = 0; dt < 2; ++dt) {
#pragma unroll
      for (int run = 0; run < 2; ++run) {
        int d0 = dt * 32 + khi * 8 + run * 16;
        int4 w = *(const int4*)(qp + d0);
        rot8((const bf16_t*)&w, ((bf16_t*)&qf[dt].v) + run * 8, pos, d0);
      }
    }
  }

  v8f zero = {0.f, 0.f, 0.f, 0.f, 0.f, 0.f, 0.f, 0.f};
  v8f sc[NT];
#pragma unroll
  for (int t = 0; t < NT; ++t) sc[t] = zero;

  // S = Q * K^T (B fragment: lane = key row, per-lane run over dh)
#pragma unroll
  for (int dt = 0; dt < 2; ++dt) {
#pragma unroll
    for (int nt = 0; nt < NT; ++nt) {
      FragBF kf;
      const bf16_t* p = Klds + (nt * 16 + lhalf) * KSTRIDE + dt * 32 + khi * 8;
      kf.q[0] = *(const int4*)p;
      kf.q[1] = *(const int4*)(p + 16);
      sc[nt] = wmma_bf16(qf[dt], kf, sc[nt]);
    }
  }

  // Exact softmax within 16-lane halves (C-matrix row layout)
  const float scale = 0.125f;  // 1/sqrt(64)
  float rinv[8];
#pragma unroll
  for (int r = 0; r < 8; ++r) {
    float mx = -1e30f;
#pragma unroll
    for (int nt = 0; nt < NT; ++nt) mx = fmaxf(mx, sc[nt][r]);
#pragma unroll
    for (int off = 8; off > 0; off >>= 1) mx = fmaxf(mx, __shfl_xor(mx, off, 32));
    float sum = 0.f;
#pragma unroll
    for (int nt = 0; nt < NT; ++nt) {
      float e = __expf((sc[nt][r] - mx) * scale);
      sc[nt][r] = e;
      sum += e;
    }
#pragma unroll
    for (int off = 8; off > 0; off >>= 1) sum += __shfl_xor(sum, off, 32);
    rinv[r] = 1.f / sum;
  }

  // Re-lay P through LDS into A-fragment form
  const int rowadd = khi * 8;
#pragma unroll
  for (int nt = 0; nt < NT; ++nt)
#pragma unroll
    for (int r = 0; r < 8; ++r)
      Plds[(r + rowadd) * VSTRIDE + nt * 16 + lhalf] = (bf16_t)sc[nt][r];
  __syncthreads();

  // O = P * V
  v8f oc[4];
#pragma unroll
  for (int dt = 0; dt < 4; ++dt) oc[dt] = zero;
#pragma unroll
  for (int kt = 0; kt < NCTX / 32; ++kt) {
    FragBF pf;
    const bf16_t* pp = Plds + lhalf * VSTRIDE + kt * 32 + khi * 8;
    pf.q[0] = *(const int4*)pp;
    pf.q[1] = *(const int4*)(pp + 16);
#pragma unroll
    for (int dt = 0; dt < 4; ++dt) {
      FragBF vf;
      const bf16_t* vp = Vlds + (dt * 16 + lhalf) * VSTRIDE + kt * 32 + khi * 8;
      vf.q[0] = *(const int4*)vp;
      vf.q[1] = *(const int4*)(vp + 16);
      oc[dt] = wmma_bf16(pf, vf, oc[dt]);
    }
  }

  // Normalize by row sums and store
#pragma unroll
  for (int dt = 0; dt < 4; ++dt)
#pragma unroll
    for (int r = 0; r < 8; ++r) {
      float v = oc[dt][r] * rinv[r];
      out[(rowbase + i0 + r + rowadd) * 512 + h * 64 + dt * 16 + lhalf] = (bf16_t)v;
    }
}

// ---------------------------------------------------------------------------
// LayerNorm family. One wave per 512-wide row; 16 values per lane.
// MODE 0: dstb[r] = LN(xa[r])
// MODE 1: dstb[r] = LN(xa[ra]+xb[ra]),  ra = (b,t,p) source of (b,p,t) row r
// MODE 2: dstf[r] = LN(xa[r]+xb[rb])*g+b + xres[r],  rb = (b,p,t) source
// ---------------------------------------------------------------------------
template<int MODE>
__global__ __launch_bounds__(256) void ln_kernel(
    const float* __restrict__ xa, const float* __restrict__ xb,
    const float* __restrict__ xres, const float* __restrict__ g,
    const float* __restrict__ bv, bf16_t* __restrict__ dstb,
    float* __restrict__ dstf, int R) {
  int wave = threadIdx.x >> 5, lane = threadIdx.x & 31;
  int r = blockIdx.x * 8 + wave;
  if (r >= R) return;
  int ra = r, rb = r;
  if (MODE == 1) {  // dst row (b,p,t) -> src row (b,t,p)  (t=64, p=256)
    int b_ = r / (256 * 64); int rem = r - b_ * (256 * 64);
    int pp = rem >> 6, tt = rem & 63;
    ra = (b_ * 64 + tt) * 256 + pp; rb = ra;
  } else if (MODE == 2) {  // dst row (b,t,p) -> xt row (b,p,t)
    int b_ = r / (64 * 256); int rem = r - b_ * (64 * 256);
    int tt = rem >> 8, pp = rem & 255;
    rb = (b_ * 256 + pp) * 64 + tt;
  }
  const float* rowa = xa + (size_t)ra * 512;
  const float* rowb = xb + (size_t)rb * 512;
  float v[16];
  int c0 = lane * 16;
#pragma unroll
  for (int q = 0; q < 4; ++q) {
    float4 fa = *(const float4*)(rowa + c0 + q * 4);
    if (MODE != 0) {
      float4 fb = *(const float4*)(rowb + c0 + q * 4);
      fa.x += fb.x; fa.y += fb.y; fa.z += fb.z; fa.w += fb.w;
    }
    v[q * 4 + 0] = fa.x; v[q * 4 + 1] = fa.y; v[q * 4 + 2] = fa.z; v[q * 4 + 3] = fa.w;
  }
  float mu = 0.f;
#pragma unroll
  for (int i = 0; i < 16; ++i) mu += v[i];
  mu = allreduce_sum(mu) * (1.f / 512.f);
  float var = 0.f;
#pragma unroll
  for (int i = 0; i < 16; ++i) { float d = v[i] - mu; var += d * d; }
  var = allreduce_sum(var) * (1.f / 512.f);
  float rs = rsqrtf(var + 1e-5f);
  if (MODE == 2) {
    const float* rowr = xres + (size_t)r * 512;
    float* drow = dstf + (size_t)r * 512;
#pragma unroll
    for (int i = 0; i < 16; ++i) {
      int c = c0 + i;
      drow[c] = (v[i] - mu) * rs * g[c] + bv[c] + rowr[c];
    }
  } else {
    bf16_t* drow = dstb + (size_t)r * 512;
#pragma unroll
    for (int i = 0; i < 16; ++i) {
      int c = c0 + i;
      drow[c] = (bf16_t)((v[i] - mu) * rs * g[c] + bv[c]);
    }
  }
}

// ---------------------------------------------------------------------------
// Host orchestration
// ---------------------------------------------------------------------------
extern "C" void kernel_launch(void* const* d_in, const int* in_sizes, int n_in,
                              void* d_out, int out_size, void* d_ws, size_t ws_size,
                              hipStream_t stream) {
  (void)in_sizes; (void)n_in; (void)out_size; (void)ws_size;
  const float* x       = (const float*)d_in[0];
  const float* ln_s_g  = (const float*)d_in[1];
  const float* ln_s_b  = (const float*)d_in[2];
  const float* w_qkv_s = (const float*)d_in[3];
  const float* w_out_s = (const float*)d_in[4];
  const float* b_out_s = (const float*)d_in[5];
  const float* ln_t_g  = (const float*)d_in[6];
  const float* ln_t_b  = (const float*)d_in[7];
  const float* w_qkv_t = (const float*)d_in[8];
  const float* w_out_t = (const float*)d_in[9];
  const float* b_out_t = (const float*)d_in[10];
  const float* ln_f_g  = (const float*)d_in[11];
  const float* ln_f_b  = (const float*)d_in[12];

  const int R = 32768;          // b*t*p = 2*64*256
  float* xcur = (float*)d_out;  // residual stream lives in d_out

  char* ws = (char*)d_ws;
  size_t off = 0;
  auto alloc = [&](size_t bytes) -> void* {
    void* p = ws + off;
    off += (bytes + 255) & ~(size_t)255;
    return p;
  };
  bf16_t* wqs  = (bf16_t*)alloc((size_t)4 * 1536 * 512 * 2);
  bf16_t* wqt  = (bf16_t*)alloc((size_t)4 * 1536 * 512 * 2);
  bf16_t* wos  = (bf16_t*)alloc((size_t)4 * 512 * 512 * 2);
  bf16_t* wot  = (bf16_t*)alloc((size_t)4 * 512 * 512 * 2);
  bf16_t* xn   = (bf16_t*)alloc((size_t)R * 512 * 2);
  bf16_t* qkv  = (bf16_t*)alloc((size_t)R * 1536 * 2);
  bf16_t* aout = (bf16_t*)alloc((size_t)R * 512 * 2);
  float*  xs   = (float*)alloc((size_t)R * 512 * 4);
  float*  xt   = (float*)alloc((size_t)R * 512 * 4);

  const size_t shm_s = ((size_t)256 * 72 + 64 * 264 + 8 * 16 * 264) * 2;  // attn<256,8>
  const size_t shm_t = ((size_t)64 * 72 + 64 * 72 + 4 * 16 * 72) * 2;     // attn<64,4>
  (void)hipFuncSetAttribute(reinterpret_cast<const void*>(attn_kernel<256, 8>),
                            hipFuncAttributeMaxDynamicSharedMemorySize, (int)shm_s);
  (void)hipFuncSetAttribute(reinterpret_cast<const void*>(attn_kernel<64, 4>),
                            hipFuncAttributeMaxDynamicSharedMemorySize, (int)shm_t);

  // Weight prep (bf16 + transpose to [N][K])
  const int tq = 512 * 1536, to = 512 * 512;
  for (int i = 0; i < 4; ++i) {
    wprep_kernel<<<(tq + 255) / 256, 256, 0, stream>>>(w_qkv_s + (size_t)i * tq, wqs + (size_t)i * tq, 512, 1536);
    wprep_kernel<<<(tq + 255) / 256, 256, 0, stream>>>(w_qkv_t + (size_t)i * tq, wqt + (size_t)i * tq, 512, 1536);
    wprep_kernel<<<(to + 255) / 256, 256, 0, stream>>>(w_out_s + (size_t)i * to, wos + (size_t)i * to, 512, 512);
    wprep_kernel<<<(to + 255) / 256, 256, 0, stream>>>(w_out_t + (size_t)i * to, wot + (size_t)i * to, 512, 512);
  }
  copy_kernel<<<(R * 512 / 4) / 256, 256, 0, stream>>>((const float4*)x, (float4*)xcur, R * 512 / 4);

  for (int i = 0; i < 4; ++i) {
    // ---- spatial branch (n = 256 over p) ----
    ln_kernel<0><<<R / 8, 256, 0, stream>>>(xcur, nullptr, nullptr,
                                            ln_s_g + i * 512, ln_s_b + i * 512, xn, nullptr, R);
    gemm_kernel<false><<<dim3(12, R / 128), 256, 0, stream>>>(
        xn, wqs + (size_t)i * tq, nullptr, qkv, nullptr, R, 1536, 512);
    attn_kernel<256, 8><<<dim3(128 * 8, 2), 256, shm_s, stream>>>(qkv, aout);
    gemm_kernel<true><<<dim3(4, R / 128), 256, 0, stream>>>(
        aout, wos + (size_t)i * to, xs, nullptr, b_out_s + i * 512, R, 512, 512);
    // ---- temporal branch (n = 64 over t) ----
    ln_kernel<1><<<R / 8, 256, 0, stream>>>(xcur, xs, nullptr,
                                            ln_t_g + i * 512, ln_t_b + i * 512, xn, nullptr, R);
    gemm_kernel<false><<<dim3(12, R / 128), 256, 0, stream>>>(
        xn, wqt + (size_t)i * tq, nullptr, qkv, nullptr, R, 1536, 512);
    attn_kernel<64, 4><<<dim3(512 * 8, 1), 128, shm_t, stream>>>(qkv, aout);
    gemm_kernel<true><<<dim3(4, R / 128), 256, 0, stream>>>(
        aout, wot + (size_t)i * to, xt, nullptr, b_out_t + i * 512, R, 512, 512);
    // ---- merge: x = LN(x_s + x_t) + x ----
    ln_kernel<2><<<R / 8, 256, 0, stream>>>(xs, xt, xcur,
                                            ln_f_g + i * 512, ln_f_b + i * 512, nullptr, xcur, R);
  }
}
